// ManualLSTM_59622736003849
// MI455X (gfx1250) — matrix operand order repaired and checked by reading, not compiled, not run on hardware
//
#include <hip/hip_runtime.h>
#include <hip/hip_bf16.h>

// ---------------------------------------------------------------------------
// LSTM (B=64, T=512, I=H=1024) for MI455X / gfx1250.
//   Phase 1: x_proj = bf16(x) @ bf16(W_ih)^T + (b_ih+b_hh)
//            LDS-staged, double-buffered WMMA GEMM (128x128 block tile),
//            staged via global_load_async_to_lds_b128 (ASYNCcnt).
//   Phase 2: 512 sequential fused step kernels:
//            h staged once into LDS; gates = x_proj[t] + h @ W_hh^T (WMMA),
//            LDS gate exchange -> sigmoid/tanh -> c,h update.
//   W_hh (8.4 MB bf16) stays resident in the 192 MB L2 across all steps.
// ---------------------------------------------------------------------------

typedef __bf16 bf16_t;
typedef __attribute__((ext_vector_type(8)))  __bf16 v8bf;
typedef __attribute__((ext_vector_type(16))) __bf16 v16bf;
typedef __attribute__((ext_vector_type(8)))  float  v8f;
typedef __attribute__((ext_vector_type(4)))  int    v4i;
typedef __attribute__((address_space(1))) v4i v4i_g;   // global int4
typedef __attribute__((address_space(3))) v4i v4i_l;   // LDS int4

#define BB 64
#define TT 512
#define II 1024
#define HH 1024
#define G4 4096   // 4*H

// ---- optional CDNA5 async global->LDS copy path ----------------------------
#if defined(__has_builtin)
#  if __has_builtin(__builtin_amdgcn_global_load_async_to_lds_b128)
#    define HAS_ASYNC_LDS 1
#  endif
#  if __has_builtin(__builtin_amdgcn_s_wait_asynccnt)
#    define HAS_WAIT_ASYNC 1
#  endif
#endif

__device__ __forceinline__ void async_copy16(void* lds_dst, const void* gsrc) {
#ifdef HAS_ASYNC_LDS
  __builtin_amdgcn_global_load_async_to_lds_b128(
      (v4i_g*)gsrc, (v4i_l*)lds_dst, 0, 0);
#else
  *(v8bf*)lds_dst = *(const v8bf*)gsrc;   // global_load_b128 + ds_store_b128
#endif
}

__device__ __forceinline__ void wait_async_copies() {
#ifdef HAS_ASYNC_LDS
#  ifdef HAS_WAIT_ASYNC
  __builtin_amdgcn_s_wait_asynccnt(0);
#  else
  asm volatile("s_wait_asynccnt 0" ::: "memory");
#  endif
#endif
}

// ---- WMMA fragment loaders (wave32, 16x16x32 bf16) -------------------------
// A (16x32, row-major source, stride lda elems):
//   lanes 0-15 : row m=lane,    K = {0..7} and {16..23}
//   lanes 16-31: row m=lane-16, K = {8..15} and {24..31}
__device__ __forceinline__ v16bf load_a_frag(const bf16_t* __restrict__ base,
                                             int lda, int lane) {
  const int m = lane & 15, half = lane >> 4;
  const bf16_t* p = base + (size_t)m * lda + half * 8;
  v8bf lo = *(const v8bf*)p;
  v8bf hi = *(const v8bf*)(p + 16);
  return __builtin_shufflevector(lo, hi, 0,1,2,3,4,5,6,7,
                                         8,9,10,11,12,13,14,15);
}

// B (32x16): source row n holds K contiguously:
//   lanes 0-15 : col n=lane,    K = {0..7} and {16..23}
//   lanes 16-31: col n=lane-16, K = {8..15} and {24..31}
__device__ __forceinline__ v16bf load_b_frag(const bf16_t* __restrict__ base,
                                             int ldb, int lane) {
  const int n = lane & 15, half = lane >> 4;
  const bf16_t* p = base + (size_t)n * ldb + half * 8;
  v8bf lo = *(const v8bf*)p;
  v8bf hi = *(const v8bf*)(p + 16);
  return __builtin_shufflevector(lo, hi, 0,1,2,3,4,5,6,7,
                                         8,9,10,11,12,13,14,15);
}

__device__ __forceinline__ v8f wmma_bf16(v16bf a, v16bf b, v8f c) {
  return __builtin_amdgcn_wmma_f32_16x16x32_bf16(
      false, a, false, b, (short)0, c, false, false);
}

__device__ __forceinline__ float sigmoidf_fast(float x) {
  return 1.0f / (1.0f + __expf(-x));
}

// ---- prep kernels ----------------------------------------------------------
__global__ void f32_to_bf16_kernel(const float* __restrict__ src,
                                   bf16_t* __restrict__ dst, size_t n) {
  size_t i = (size_t)blockIdx.x * blockDim.x + threadIdx.x;
  size_t stride = (size_t)gridDim.x * blockDim.x;
  for (; i < n; i += stride) dst[i] = (bf16_t)src[i];
}

__global__ void bias_sum_kernel(const float* __restrict__ a,
                                const float* __restrict__ b,
                                float* __restrict__ o, int n) {
  int i = blockIdx.x * blockDim.x + threadIdx.x;
  if (i < n) o[i] = a[i] + b[i];
}

__global__ void init_state_kernel(bf16_t* __restrict__ h0,
                                  float* __restrict__ c0, int n) {
  int i = blockIdx.x * blockDim.x + threadIdx.x;
  if (i < n) { h0[i] = (bf16_t)0.0f; c0[i] = 0.0f; }
}

// ---- Phase 1: x_proj GEMM (M=32768, N=4096, K=1024) ------------------------
// block = 256 threads (8 waves); block tile 128(M) x 128(N), K-step 32.
// LDS double-buffered A/B tiles (rows padded 32->40 elems for bank spread).
// wave tile: 32(M) x 64(N) = 2x4 WMMA tiles.
#define KPAD 40
__global__ void __launch_bounds__(256)
xproj_gemm_kernel(const bf16_t* __restrict__ xb,     // [32768,1024]
                  const bf16_t* __restrict__ wih,    // [4096,1024] row-major
                  const float*  __restrict__ bias,   // [4096]
                  bf16_t* __restrict__ xproj) {      // [32768,4096]
  __shared__ bf16_t a_sm[2][128][KPAD];              // 2 x 10 KB
  __shared__ bf16_t b_sm[2][128][KPAD];              // 2 x 10 KB

  const int lane = threadIdx.x & 31;
  const int wave = threadIdx.x >> 5;
  const int tid  = threadIdx.x;
  const int m_blk = blockIdx.x * 128;
  const int n_blk = blockIdx.y * 128;
  const int wm = (wave >> 1) * 32;   // wave M offset in block tile
  const int wn = (wave & 1) * 64;    // wave N offset in block tile

  // stage one 128x32 K-slice of A and B into LDS buffer `buf`
  auto stage = [&](int buf, int k0) {
    // 512 granules of 16B each per matrix; 2 per thread per matrix
    for (int g = tid; g < 512; g += 256) {
      const int row = g >> 2, kk = (g & 3) * 8;
      async_copy16(&a_sm[buf][row][kk],
                   xb + (size_t)(m_blk + row) * II + k0 + kk);
      async_copy16(&b_sm[buf][row][kk],
                   wih + (size_t)(n_blk + row) * II + k0 + kk);
    }
  };

  v8f acc[2][4] = {};

  stage(0, 0);
  const int NK = II / 32;
  for (int kt = 0; kt < NK; ++kt) {
    const int cur = kt & 1;
    wait_async_copies();
    __syncthreads();                       // buf[cur] visible, prev reads done
    if (kt + 1 < NK) {
      __builtin_prefetch(xb + (size_t)(m_blk + tid / 2) * II + (kt + 2) * 32, 0, 3);
      stage(cur ^ 1, (kt + 1) * 32);
    }
    v16bf a0 = load_a_frag(&a_sm[cur][wm     ][0], KPAD, lane);
    v16bf a1 = load_a_frag(&a_sm[cur][wm + 16][0], KPAD, lane);
#pragma unroll
    for (int nt = 0; nt < 4; ++nt) {
      v16bf b = load_b_frag(&b_sm[cur][wn + nt * 16][0], KPAD, lane);
      acc[0][nt] = wmma_bf16(a0, b, acc[0][nt]);
      acc[1][nt] = wmma_bf16(a1, b, acc[1][nt]);
    }
  }

  // epilogue: VGPR r of a D tile holds (m = r + 8*half, n = lane&15)
  const int half = lane >> 4, nl = lane & 15;
#pragma unroll
  for (int mt = 0; mt < 2; ++mt)
#pragma unroll
    for (int nt = 0; nt < 4; ++nt) {
      const int n = n_blk + wn + nt * 16 + nl;
      const float bv = bias[n];
#pragma unroll
      for (int r = 0; r < 8; ++r) {
        const int m = m_blk + wm + mt * 16 + r + half * 8;
        xproj[(size_t)m * G4 + n] = (bf16_t)(acc[mt][nt][r] + bv);
      }
    }
}

// ---- Phase 2: one fused LSTM step ------------------------------------------
// grid = H/64 = 16 blocks; block = 256 threads (8 waves).
// Dynamic LDS (132 KB of the 320 KB/WGP):
//   during K-loop : h_lds [64][1032] bf16 (padded rows, bank-spread)
//   after K-loop  : gbuf  [4][64][64] f32 overlaid on same memory
#define HPAD 1032
__global__ void __launch_bounds__(256)
lstm_step_kernel(const bf16_t* __restrict__ xproj,   // [B*T, 4096] bf16
                 const bf16_t* __restrict__ whh,     // [4096,1024] bf16 rowmaj
                 const bf16_t* __restrict__ hb_in,   // [64,1024] bf16
                 bf16_t* __restrict__ hb_out,        // [64,1024] bf16
                 float*  __restrict__ c_st,          // [64,1024] f32
                 float*  __restrict__ h_out,         // [64,1024] f32
                 int t) {
  extern __shared__ __align__(16) char smem[];
  bf16_t* h_lds = (bf16_t*)smem;                     // [64][HPAD]
  float*  gbuf  = (float*)smem;                      // [4][64][64] (overlay)

  const int lane = threadIdx.x & 31;
  const int wave = threadIdx.x >> 5;
  const int tid  = threadIdx.x;
  const int g    = wave >> 1;       // gate 0..3 (i,f,g,o)
  const int nh   = wave & 1;        // 32-col half of the block's 64 cols
  const int c0   = blockIdx.x * 64; // column base within H
  const int nb0  = c0 + nh * 32;

  // stage all of h (A matrix) into LDS once: 8192 granules of 16B
  for (int gr = tid; gr < 8192; gr += 256) {
    const int row = gr >> 7, kk = (gr & 127) * 8;
    async_copy16(h_lds + (size_t)row * HPAD + kk,
                 hb_in + (size_t)row * HH + kk);
  }
  wait_async_copies();
  __syncthreads();

  v8f acc[4][2] = {};

  for (int k0 = 0; k0 < HH; k0 += 32) {
    const bf16_t* wrow = whh + (size_t)(g * HH + nb0) * HH + k0;
    __builtin_prefetch(wrow + 64, 0, 3);
    v16bf b0 = load_b_frag(wrow, HH, lane);
    v16bf b1 = load_b_frag(wrow + (size_t)16 * HH, HH, lane);
#pragma unroll
    for (int mt = 0; mt < 4; ++mt) {
      v16bf a = load_a_frag(h_lds + (size_t)(mt * 16) * HPAD + k0, HPAD, lane);
      acc[mt][0] = wmma_bf16(a, b0, acc[mt][0]);
      acc[mt][1] = wmma_bf16(a, b1, acc[mt][1]);
    }
  }

  __syncthreads();   // all h_lds reads done; safe to overlay gbuf

  const int half = lane >> 4, nl = lane & 15;
#pragma unroll
  for (int mt = 0; mt < 4; ++mt)
#pragma unroll
    for (int nt = 0; nt < 2; ++nt)
#pragma unroll
      for (int r = 0; r < 8; ++r)
        gbuf[((g * 64) + mt * 16 + r + half * 8) * 64 +
             nh * 32 + nt * 16 + nl] = acc[mt][nt][r];

  __syncthreads();

  // elementwise gate update: 16 (b,col) pairs per thread, coalesced in col
  for (int p = 0; p < 16; ++p) {
    const int linear = p * 256 + tid;
    const int b   = linear >> 6;
    const int col = linear & 63;
    const size_t xrow = ((size_t)b * TT + t) * (size_t)G4 + (c0 + col);

    float iv = gbuf[((0 * 64) + b) * 64 + col] + (float)xproj[xrow + 0 * HH];
    float fv = gbuf[((1 * 64) + b) * 64 + col] + (float)xproj[xrow + 1 * HH];
    float gv = gbuf[((2 * 64) + b) * 64 + col] + (float)xproj[xrow + 2 * HH];
    float ov = gbuf[((3 * 64) + b) * 64 + col] + (float)xproj[xrow + 3 * HH];

    iv = sigmoidf_fast(iv);
    fv = sigmoidf_fast(fv);
    gv = tanhf(gv);
    ov = sigmoidf_fast(ov);

    const size_t idx = (size_t)b * HH + c0 + col;
    const float cn = fv * c_st[idx] + iv * gv;
    const float hn = ov * tanhf(cn);
    c_st[idx]   = cn;
    hb_out[idx] = (bf16_t)hn;
    h_out[idx]  = hn;
  }
}

// ---------------------------------------------------------------------------
extern "C" void kernel_launch(void* const* d_in, const int* in_sizes, int n_in,
                              void* d_out, int out_size, void* d_ws, size_t ws_size,
                              hipStream_t stream) {
  const float* x    = (const float*)d_in[0];  // (64,512,1024)
  const float* W_ih = (const float*)d_in[1];  // (4096,1024)
  const float* W_hh = (const float*)d_in[2];  // (4096,1024)
  const float* b_ih = (const float*)d_in[3];  // (4096)
  const float* b_hh = (const float*)d_in[4];  // (4096)

  // workspace carve-up (~352 MB total)
  char* ws = (char*)d_ws;
  size_t off = 0;
  auto carve = [&](size_t bytes) -> void* {
    off = (off + 255) & ~(size_t)255;
    void* p = ws + off;
    off += bytes;
    return p;
  };
  bf16_t* xb    = (bf16_t*)carve((size_t)BB * TT * II * 2);   //  67 MB
  bf16_t* wihb  = (bf16_t*)carve((size_t)G4 * II * 2);        // 8.4 MB
  bf16_t* whhb  = (bf16_t*)carve((size_t)G4 * HH * 2);        // 8.4 MB
  float*  bias  = (float* )carve((size_t)G4 * 4);
  bf16_t* xproj = (bf16_t*)carve((size_t)BB * TT * G4 * 2);   // 268 MB
  bf16_t* hb0   = (bf16_t*)carve((size_t)BB * HH * 2);
  bf16_t* hb1   = (bf16_t*)carve((size_t)BB * HH * 2);
  float*  c_st  = (float* )carve((size_t)BB * HH * 4);
  float*  h_f32 = (float* )carve((size_t)BB * HH * 4);
  (void)ws_size; (void)in_sizes; (void)n_in; (void)out_size;

  // --- prep: bf16 conversions, bias sum, state init ---
  f32_to_bf16_kernel<<<4096, 256, 0, stream>>>(x, xb, (size_t)BB * TT * II);
  f32_to_bf16_kernel<<<2048, 256, 0, stream>>>(W_ih, wihb, (size_t)G4 * II);
  f32_to_bf16_kernel<<<2048, 256, 0, stream>>>(W_hh, whhb, (size_t)G4 * HH);
  bias_sum_kernel<<<G4 / 256, 256, 0, stream>>>(b_ih, b_hh, bias, G4);
  init_state_kernel<<<(BB * HH) / 256, 256, 0, stream>>>(hb0, c_st, BB * HH);

  // --- phase 1: x_proj = x @ W_ih^T + bias ---
  dim3 ggrid((BB * TT) / 128, G4 / 128);
  xproj_gemm_kernel<<<ggrid, 256, 0, stream>>>(xb, wihb, bias, xproj);

  // --- phase 2: 512 sequential fused steps, double-buffered h ---
  const size_t step_lds = (size_t)BB * HPAD * sizeof(bf16_t);  // 132 KB
  for (int t = 0; t < TT; ++t) {
    bf16_t* hin  = (t & 1) ? hb1 : hb0;
    bf16_t* hout = (t & 1) ? hb0 : hb1;
    float*  hw   = (t == TT - 1) ? (float*)d_out : h_f32;
    lstm_step_kernel<<<HH / 64, 256, step_lds, stream>>>(xproj, whhb, hin,
                                                         hout, c_st, hw, t);
  }
}
